// LearnedSparseAttention_46059229283001
// MI455X (gfx1250) — matrix operand order repaired and checked by reading, not compiled, not run on hardware
//
#include <hip/hip_runtime.h>
#include <hip/hip_bf16.h>

typedef __attribute__((ext_vector_type(16))) _Float16 v16h;
typedef __attribute__((ext_vector_type(8)))  _Float16 v8h;
typedef __attribute__((ext_vector_type(8)))  float    v8f;

union HV { v16h v; v8h h[2]; };

#define WMMA_F16(a, b, c) \
  __builtin_amdgcn_wmma_f32_16x16x32_f16(false, (a), false, (b), (short)0, (c), false, false)

#define USE_ASYNC_LDS 1

static __device__ __forceinline__ void wait_dscnt0() {
  asm volatile("s_wait_dscnt 0" ::: "memory");
}

#if USE_ASYNC_LDS
// GLOBAL_LOAD_ASYNC_TO_LDS_B128: vdst = LDS byte offset (u32 VGPR),
// vaddr = 64-bit global address. Tracked by ASYNCcnt.
static __device__ __forceinline__ void async_ld_b128(unsigned lds_off, const void* g) {
  asm volatile("global_load_async_to_lds_b128 %0, %1, off"
               :: "v"(lds_off), "v"((unsigned long long)(uintptr_t)g)
               : "memory");
}
static __device__ __forceinline__ void wait_async0() {
  asm volatile("s_wait_asynccnt 0" ::: "memory");
}
#else
static __device__ __forceinline__ void wait_async0() {}
#endif

// constants
#define BB   2
#define SS   2048
#define DD   1024
#define HH   16
#define HKV  4
#define HD   64
#define SD   16
#define MTOT (BB * SS)   // 4096

// ---------------------------------------------------------------------------
// f32 -> f16 conversion (vectorized 4/thread)
// ---------------------------------------------------------------------------
__global__ void cvt_f32_f16_kernel(const float* __restrict__ s,
                                   _Float16* __restrict__ d, int n) {
  int i = (blockIdx.x * blockDim.x + threadIdx.x) * 4;
  if (i + 3 < n) {
    float4 f = *(const float4*)(s + i);
    _Float16 h0 = (_Float16)f.x, h1 = (_Float16)f.y;
    _Float16 h2 = (_Float16)f.z, h3 = (_Float16)f.w;
    d[i] = h0; d[i + 1] = h1; d[i + 2] = h2; d[i + 3] = h3;
  } else {
    for (int t = i; t < n; ++t) d[t] = (_Float16)s[t];
  }
}

// ---------------------------------------------------------------------------
// GEMM: C[M,N](f32) = A[M,K](f16) @ W[N,K](f16)^T (+bias)
// 128 threads (4 waves), 64x64 tile, K-step 32, double-buffered LDS with
// async global->LDS b128 copies.
// ---------------------------------------------------------------------------
static __device__ __forceinline__ void tile_copy(
    const _Float16* __restrict__ A, const _Float16* __restrict__ W,
    _Float16 (*__restrict__ As)[32], _Float16 (*__restrict__ Bs)[32],
    int m0, int n0, int k0, int K, int tid) {
  // 64 rows x 32 halfs for each of As/Bs; 4x 16B chunks per row; 2 per thread
#pragma unroll
  for (int c = 0; c < 2; ++c) {
    int id  = tid + c * 128;          // 0..255
    int row = id >> 2;
    int seg = (id & 3) * 8;
    const _Float16* ga = A + (size_t)(m0 + row) * K + k0 + seg;
    const _Float16* gb = W + (size_t)(n0 + row) * K + k0 + seg;
#if USE_ASYNC_LDS
    async_ld_b128((unsigned)(uintptr_t)&As[row][seg], ga);
    async_ld_b128((unsigned)(uintptr_t)&Bs[row][seg], gb);
#else
    *(v8h*)&As[row][seg] = *(const v8h*)ga;
    *(v8h*)&Bs[row][seg] = *(const v8h*)gb;
#endif
  }
}

__global__ __launch_bounds__(128)
void gemm_h16_kernel(const _Float16* __restrict__ A, const _Float16* __restrict__ W,
                     const float* __restrict__ bias, float* __restrict__ C,
                     int M, int N, int K) {
  __shared__ _Float16 AsBuf[2][64][32];
  __shared__ _Float16 BsBuf[2][64][32];

  const int tid  = threadIdx.x;
  const int lane = tid & 31;
  const int wave = tid >> 5;
  const int m0 = blockIdx.x * 64;
  const int n0 = blockIdx.y * 64;

  const int mrow  = wave * 16 + (lane & 15);
  const int kbase = (lane >> 4) * 8;
  const int nl    = lane & 15;
  const int kgrp  = (lane >> 4) * 16;

  v8f acc[4] = {};
  const int nk = K >> 5;

  tile_copy(A, W, AsBuf[0], BsBuf[0], m0, n0, 0, K, tid);

  for (int t = 0; t < nk; ++t) {
    wait_async0();
    __syncthreads();
    if (t + 1 < nk)
      tile_copy(A, W, AsBuf[(t + 1) & 1], BsBuf[(t + 1) & 1],
                m0, n0, (t + 1) << 5, K, tid);
    const int cur = t & 1;

    HV a;
    a.h[0] = *(const v8h*)&AsBuf[cur][mrow][kbase];        // K = kbase..+7
    a.h[1] = *(const v8h*)&AsBuf[cur][mrow][16 + kbase];   // K = kbase+16..+23
#pragma unroll
    for (int f = 0; f < 4; ++f) {
      HV b;
      b.h[0] = *(const v8h*)&BsBuf[cur][f * 16 + nl][kbase];
      b.h[1] = *(const v8h*)&BsBuf[cur][f * 16 + nl][16 + kbase];
      acc[f] = WMMA_F16(a.v, b.v, acc[f]);
    }
    __syncthreads();
  }

  const int mbase = (lane >> 4) * 8;
#pragma unroll
  for (int f = 0; f < 4; ++f) {
#pragma unroll
    for (int i = 0; i < 8; ++i) {
      int row = m0 + wave * 16 + mbase + i;
      int col = n0 + f * 16 + nl;
      float val = acc[f][i];
      if (bias) val += bias[col];
      C[(size_t)row * N + col] = val;
    }
  }
}

// ---------------------------------------------------------------------------
// RoPE (rotate-half), in place on [MTOT, nheads*64]
// ---------------------------------------------------------------------------
__global__ void rope_kernel(float* __restrict__ qk, int nheads, int ld) {
  int idx   = blockIdx.x * blockDim.x + threadIdx.x;
  int total = MTOT * nheads * 32;
  if (idx >= total) return;
  int d = idx & 31;
  int h = (idx >> 5) % nheads;
  int m = idx / (32 * nheads);
  int s = m & (SS - 1);
  float theta = __expf((float)d * (-9.210340371976184f / 32.0f)); // 10000^(-d/32)
  float ang = (float)s * theta;
  float sn, cs;
  __sincosf(ang, &sn, &cs);
  float* base = qk + (size_t)m * ld + h * 64;
  float x1 = base[d];
  float x2 = base[d + 32];
  base[d]      = x1 * cs - x2 * sn;
  base[d + 32] = x2 * cs + x1 * sn;
}

// ---------------------------------------------------------------------------
// Sparsity projection: out[b,h,s,0:16] = qk_head[s,0:64] @ Ws^T + bs
// ---------------------------------------------------------------------------
__global__ __launch_bounds__(32)
void spproj_kernel(const float* __restrict__ qk, int ld, int nheads,
                   const float* __restrict__ Ws, const float* __restrict__ bs,
                   float* __restrict__ outp) {
  const int lane  = threadIdx.x & 31;
  const int m0    = blockIdx.x * 16;
  const int h     = blockIdx.y;
  const int kbase = (lane >> 4) * 8;
  const int mrow  = m0 + (lane & 15);

  const float* src = qk + (size_t)mrow * ld + h * 64;
  v16h a0, a1;
#pragma unroll
  for (int i = 0; i < 8; ++i)  { a0[i] = (_Float16)src[kbase + i];
                                 a1[i] = (_Float16)src[32 + kbase + i]; }
#pragma unroll
  for (int i = 8; i < 16; ++i) { a0[i] = (_Float16)src[8 + kbase + i];
                                 a1[i] = (_Float16)src[40 + kbase + i]; }

  const int nl   = lane & 15;
  const int kgrp = (lane >> 4) * 16;
  v16h b0, b1;
#pragma unroll
  for (int i = 0; i < 16; ++i) {
    b0[i] = (_Float16)Ws[nl * 64 + (i < 8 ? kbase + i : 8 + kbase + i)];
    b1[i] = (_Float16)Ws[nl * 64 + 32 + (i < 8 ? kbase + i : 8 + kbase + i)];
  }

  v8f c = {};
  c = WMMA_F16(a1, b1, c);
  c = WMMA_F16(a0, b0, c);

  const int mbase = (lane >> 4) * 8;
  float bias = bs[nl];
#pragma unroll
  for (int i = 0; i < 8; ++i) {
    int row = m0 + mbase + i;
    int bb  = row >> 11;
    int s   = row & (SS - 1);
    outp[(((size_t)bb * nheads + h) * SS + s) * SD + nl] = c[i] + bias;
  }
}

// ---------------------------------------------------------------------------
// Gated causal flash attention (output stored as f16 for the Wo GEMM).
// grid = (S/128, B*H), block = 256 (8 waves), wave-per-16-query-rows.
// ---------------------------------------------------------------------------
__global__ __launch_bounds__(256)
void sparse_attn_kernel(const float* __restrict__ q, const float* __restrict__ k,
                        const float* __restrict__ v, const float* __restrict__ qsp,
                        const float* __restrict__ ksp, _Float16* __restrict__ outp) {
  __shared__ _Float16 pbuf[8][16][24];   // 48B rows -> 16B-aligned v8h reads

  const int lane = threadIdx.x & 31;
  const int wave = threadIdx.x >> 5;
  const int qt   = blockIdx.x * 8 + wave;
  const int bh   = blockIdx.y;
  const int b    = bh >> 4;
  const int h    = bh & 15;
  const int hkv  = h >> 2;

  const int hf    = lane >> 4;
  const int kbase = hf * 8;
  const int nl    = lane & 15;
  const int kgrp  = hf * 16;
  const int q0    = qt * 16;
  const int ml    = lane & 15;

  v16h aq0, aq1, asp;
  {
    const float* src = q + ((size_t)(b * SS + q0 + ml)) * DD + h * 64;
#pragma unroll
    for (int i = 0; i < 8; ++i)  { aq0[i] = (_Float16)src[kbase + i];
                                   aq1[i] = (_Float16)src[32 + kbase + i]; }
#pragma unroll
    for (int i = 8; i < 16; ++i) { aq0[i] = (_Float16)src[8 + kbase + i];
                                   aq1[i] = (_Float16)src[40 + kbase + i]; }
    const float* ssrc = qsp + (((size_t)(b * HH + h)) * SS + q0 + ml) * SD;
#pragma unroll
    for (int i = 0; i < 8; ++i)  asp[i] = (_Float16)ssrc[kbase + i];
#pragma unroll
    for (int i = 8; i < 16; ++i) asp[i] = (_Float16)0.0f;
  }

  v8f o[4] = {};
  float mrun[8], lrun[8];
#pragma unroll
  for (int i = 0; i < 8; ++i) { mrun[i] = -1e30f; lrun[i] = 0.0f; }

  for (int j = 0; j <= qt; ++j) {
    const int kr = j * 16;

    v16h bk0, bk1, bsp;
    const float* ksrc  = k + ((size_t)(b * SS + kr + nl)) * (HKV * HD) + hkv * 64;
    const float* kssrc = ksp + (((size_t)(b * HKV + hkv)) * SS + kr + nl) * SD;
#pragma unroll
    for (int i = 0; i < 16; ++i) {
      bk0[i] = (_Float16)ksrc[(i < 8 ? kbase + i : 8 + kbase + i)];
      bk1[i] = (_Float16)ksrc[32 + (i < 8 ? kbase + i : 8 + kbase + i)];
      bsp[i] = (i >= 8) ? (_Float16)0.0f : (_Float16)kssrc[kbase + i];
    }

    v8f sacc = {};
    sacc = WMMA_F16(aq0, bk0, sacc);
    sacc = WMMA_F16(aq1, bk1, sacc);
    v8f spacc = {};
    spacc = WMMA_F16(asp, bsp, spacc);

    float sv[8];
#pragma unroll
    for (int i = 0; i < 8; ++i) {
      float gate = 1.0f / (1.0f + __expf(-spacc[i] * 0.25f)); // /sqrt(SD)
      float val  = sacc[i] * 0.125f * gate;                   // /sqrt(HD)
      int qrow = q0 + hf * 8 + i;
      int kcol = kr + nl;
      if (kcol > qrow) val = -1e30f;
      sv[i] = val;
    }

    float rmax[8];
#pragma unroll
    for (int i = 0; i < 8; ++i) rmax[i] = sv[i];
#pragma unroll
    for (int msk = 1; msk < 16; msk <<= 1)
#pragma unroll
      for (int i = 0; i < 8; ++i)
        rmax[i] = fmaxf(rmax[i], __shfl_xor(rmax[i], msk, 32));

    float pv[8];
#pragma unroll
    for (int i = 0; i < 8; ++i) {
      float mnew  = fmaxf(mrun[i], rmax[i]);
      float alpha = __expf(mrun[i] - mnew);
      pv[i]   = __expf(sv[i] - mnew);
      mrun[i] = mnew;
      lrun[i] *= alpha;
      o[0][i] *= alpha; o[1][i] *= alpha; o[2][i] *= alpha; o[3][i] *= alpha;
    }
    float rsum[8];
#pragma unroll
    for (int i = 0; i < 8; ++i) rsum[i] = pv[i];
#pragma unroll
    for (int msk = 1; msk < 16; msk <<= 1)
#pragma unroll
      for (int i = 0; i < 8; ++i)
        rsum[i] += __shfl_xor(rsum[i], msk, 32);
#pragma unroll
    for (int i = 0; i < 8; ++i) lrun[i] += rsum[i];

    // relayout p (C layout -> A layout) via per-wave LDS (DS ops wave-ordered)
#pragma unroll
    for (int i = 0; i < 8; ++i) pbuf[wave][hf * 8 + i][nl] = (_Float16)pv[i];
    wait_dscnt0();
    HV ap;
    ap.h[0] = *(const v8h*)&pbuf[wave][ml][kbase];  // K = kbase..kbase+7
#pragma unroll
    for (int i = 8; i < 16; ++i) ap.v[i] = (_Float16)0.0f;

#pragma unroll
    for (int f = 0; f < 4; ++f) {
      v16h bv;
#pragma unroll
      for (int i = 0; i < 16; ++i)
        bv[i] = (i >= 8) ? (_Float16)0.0f
                   : (_Float16)v[((size_t)(b * SS + kr + kbase + i)) * (HKV * HD)
                                 + hkv * 64 + f * 16 + nl];
      o[f] = WMMA_F16(ap.v, bv, o[f]);
    }
  }

#pragma unroll
  for (int f = 0; f < 4; ++f) {
#pragma unroll
    for (int i = 0; i < 8; ++i) {
      int qrow = q0 + hf * 8 + i;
      outp[((size_t)(b * SS + qrow)) * DD + h * 64 + f * 16 + nl] =
          (_Float16)(o[f][i] / lrun[i]);
    }
  }
}

// ---------------------------------------------------------------------------
extern "C" void kernel_launch(void* const* d_in, const int* in_sizes, int n_in,
                              void* d_out, int out_size, void* d_ws, size_t ws_size,
                              hipStream_t stream) {
  (void)in_sizes; (void)n_in; (void)out_size; (void)ws_size;
  const float* x  = (const float*)d_in[0];
  const float* Wq = (const float*)d_in[1];
  const float* Wk = (const float*)d_in[2];
  const float* Wv = (const float*)d_in[3];
  const float* Wo = (const float*)d_in[4];
  const float* bo = (const float*)d_in[5];
  const float* Ws = (const float*)d_in[6];
  const float* bs = (const float*)d_in[7];
  float* out = (float*)d_out;

  char* w = (char*)d_ws;
  float* qb   = (float*)w;                 w += (size_t)MTOT * DD * 4;        // 16MB
  float* kb   = (float*)w;                 w += (size_t)MTOT * HKV * HD * 4;  // 4MB
  float* vb   = (float*)w;                 w += (size_t)MTOT * HKV * HD * 4;  // 4MB
  float* qsp  = (float*)w;                 w += (size_t)BB * HH * SS * SD * 4;  // 4MB
  float* kspb = (float*)w;                 w += (size_t)BB * HKV * SS * SD * 4; // 1MB
  _Float16* x16    = (_Float16*)w;         w += (size_t)MTOT * DD * 2;        // 8MB
  _Float16* wq16   = (_Float16*)w;         w += (size_t)DD * DD * 2;          // 2MB
  _Float16* wk16   = (_Float16*)w;         w += (size_t)HKV * HD * DD * 2;    // .5MB
  _Float16* wv16   = (_Float16*)w;         w += (size_t)HKV * HD * DD * 2;    // .5MB
  _Float16* wo16   = (_Float16*)w;         w += (size_t)DD * DD * 2;          // 2MB
  _Float16* attn16 = (_Float16*)w;         w += (size_t)MTOT * DD * 2;        // 8MB

  // f32 -> f16 staging (x and all GEMM weights)
  const int CT = 256;
  cvt_f32_f16_kernel<<<(MTOT * DD / 4 + CT - 1) / CT, CT, 0, stream>>>(x, x16, MTOT * DD);
  cvt_f32_f16_kernel<<<(DD * DD / 4 + CT - 1) / CT, CT, 0, stream>>>(Wq, wq16, DD * DD);
  cvt_f32_f16_kernel<<<(HKV * HD * DD / 4 + CT - 1) / CT, CT, 0, stream>>>(Wk, wk16, HKV * HD * DD);
  cvt_f32_f16_kernel<<<(HKV * HD * DD / 4 + CT - 1) / CT, CT, 0, stream>>>(Wv, wv16, HKV * HD * DD);
  cvt_f32_f16_kernel<<<(DD * DD / 4 + CT - 1) / CT, CT, 0, stream>>>(Wo, wo16, DD * DD);

  // Q/K/V projections
  gemm_h16_kernel<<<dim3(MTOT / 64, DD / 64), 128, 0, stream>>>(
      x16, wq16, nullptr, qb, MTOT, DD, DD);
  gemm_h16_kernel<<<dim3(MTOT / 64, (HKV * HD) / 64), 128, 0, stream>>>(
      x16, wk16, nullptr, kb, MTOT, HKV * HD, DD);
  gemm_h16_kernel<<<dim3(MTOT / 64, (HKV * HD) / 64), 128, 0, stream>>>(
      x16, wv16, nullptr, vb, MTOT, HKV * HD, DD);

  // RoPE
  rope_kernel<<<(MTOT * HH * 32 + 255) / 256, 256, 0, stream>>>(qb, HH, DD);
  rope_kernel<<<(MTOT * HKV * 32 + 255) / 256, 256, 0, stream>>>(kb, HKV, HKV * HD);

  // gate projections
  spproj_kernel<<<dim3(MTOT / 16, HH), 32, 0, stream>>>(qb, DD, HH, Ws, bs, qsp);
  spproj_kernel<<<dim3(MTOT / 16, HKV), 32, 0, stream>>>(kb, HKV * HD, HKV, Ws, bs, kspb);

  // gated causal flash attention -> f16
  sparse_attn_kernel<<<dim3(SS / 128, BB * HH), 256, 0, stream>>>(
      qb, kb, vb, qsp, kspb, attn16);

  // output projection with bias
  gemm_h16_kernel<<<dim3(MTOT / 64, DD / 64), 128, 0, stream>>>(
      attn16, wo16, bo, out, MTOT, DD, DD);
}
